// HybridClassifier_65481071404621
// MI455X (gfx1250) — matrix operand order repaired and checked by reading, not compiled, not run on hardware
//
#include <hip/hip_runtime.h>
#include <hip/hip_fp16.h>
#include <math.h>

// CDNA5 WMMA fragment types (wave32): 16 halves (A/B), 8 floats (C/D)
typedef __attribute__((ext_vector_type(16))) _Float16 v16h;
typedef __attribute__((ext_vector_type(8)))  float    v8f;

#define BN_EPS 1e-5f

// ---- stats workspace layout (floats) ----
// [0:16)  sum1   [16:32)  sumsq1  [32:48)   scale1  [48:64)   shift1
// [64:96) sum2   [96:128) sumsq2  [128:160) scale2  [160:192) shift2

// A-matrix (16x32 f16) per-lane K index for fragment element i (ISA 7.12.2):
// lanes 0-15: V0-3 hold K=0..7, V4-7 hold K=16..23; lanes 16-31: +8
__device__ __forceinline__ constexpr int a_k_of(int i, int half) {
    int v = i >> 1;
    int base = (v < 4) ? (2 * v) : (16 + 2 * (v - 4));
    return base + (i & 1) + 8 * half;
}

// conv1 LDS patch layout: [c(4)][row(17)][col(33)], K = c*9 + ky*3 + kx.
// Channel 3 is a zero pad region so K=27..31 reads 0 with no branch.
__device__ __forceinline__ constexpr int c1_delta(int k) {
    return (k / 9) * (17 * 33) + ((k % 9) / 3) * 33 + ((k % 9) % 3);
}

// conv2 LDS patch layout: [row(10)][col(9)][ch(16)], K = (ky*3+kx)*16 + c.
__device__ __forceinline__ constexpr int c2_delta(int k) {
    int rpos = k >> 4, cch = k & 15;
    return ((rpos / 3) * 9 + (rpos % 3)) * 16 + cch;
}

// ---------------------------------------------------------------------------
// conv1: block = 8 waves covering an 8(row) x 16(col) conv1-pixel region of
// one image. Wave (wy=wave>>2, wx=wave&3) owns a 4x4 tile = four 2x2 pool
// windows; A-row m: w=m>>2, iy=(m>>1)&1, ix=m&1.
// ---------------------------------------------------------------------------
__device__ __forceinline__ void conv1_stage(
    const float* __restrict__ x, const float* __restrict__ w1,
    _Float16* patch, _Float16* bfrag, int b, int by, int bx)
{
    const int inY0 = 16 * by - 1;   // 2*(8*by) - 1
    const int inX0 = 32 * bx - 1;   // 2*(16*bx) - 1
    for (int idx = threadIdx.x; idx < 4 * 17 * 33; idx += 256) {
        int c = idx / 561, rem = idx % 561;
        int yy = rem / 33, xx = rem % 33;
        float v = 0.0f;
        if (c < 3) {
            int gy = inY0 + yy, gx = inX0 + xx;
            if ((unsigned)gy < 224u && (unsigned)gx < 224u)
                v = x[((b * 3 + c) * 224 + gy) * 224 + gx];
        }
        patch[idx] = (_Float16)v;
    }
    // fragment-ordered B: bfrag[lane*16 + i], lane = half*16 + n
    for (int idx = threadIdx.x; idx < 512; idx += 256) {
        int lane = idx >> 4, i = idx & 15;
        int lr = lane & 15, hf = lane >> 4;
        int kb = (i & 7) + 8 * hf + 16 * (i >> 3);
        bfrag[idx] = (_Float16)((kb < 27) ? w1[lr * 27 + kb] : 0.0f);
    }
    __syncthreads();
}

__device__ __forceinline__ v8f conv1_block_tile(
    const _Float16* patch, const _Float16* bfrag, int lane, int wave)
{
    const int half = lane >> 4;
    const int lr   = lane & 15;
    const int wy = wave >> 2, wx = wave & 3;
    const int wq = lr >> 2;
    const int oyr = 4 * wy + 2 * (wq >> 1) + ((lr >> 1) & 1);
    const int oxr = 4 * wx + 2 * (wq & 1) + (lr & 1);
    const int pixbase = 2 * oyr * 33 + 2 * oxr;
    v16h a;
    #pragma unroll
    for (int i = 0; i < 16; ++i) {
        const int d0 = c1_delta(a_k_of(i, 0));      // compile-time
        const int d1 = c1_delta(a_k_of(i, 0) + 8);  // compile-time
        a[i] = patch[pixbase + (half ? d1 : d0)];   // ds_load_u16, branch-free
    }
    v16h bm = *(const v16h*)&bfrag[lane * 16];       // 2x ds_load_b128
    v8f c = {};
    return __builtin_amdgcn_wmma_f32_16x16x32_f16(false, a, false, bm,
                                                  (short)0, c, false, false);
}

__global__ void k_zero_stats(float* __restrict__ stats) {
    stats[threadIdx.x] = 0.0f;
}

// Pass 1a: conv1 raw (+bias) -> per-channel sum/sumsq
__global__ void k_conv1_stats(const float* __restrict__ x, const float* __restrict__ w1,
                              const float* __restrict__ b1, float* __restrict__ stats)
{
    __shared__ __align__(16) _Float16 patch[4 * 17 * 33];
    __shared__ __align__(16) _Float16 bfrag[512];
    __shared__ float ssum[16], ssq[16];
    if (threadIdx.x < 16) { ssum[threadIdx.x] = 0.f; ssq[threadIdx.x] = 0.f; }
    const int blk = blockIdx.x;           // 128 * 14 * 7 blocks
    const int b = blk / 98, r0 = blk % 98;
    conv1_stage(x, w1, patch, bfrag, b, r0 / 7, r0 % 7);
    const int lane = threadIdx.x & 31, wave = threadIdx.x >> 5;
    v8f c = conv1_block_tile(patch, bfrag, lane, wave);
    const int lr = lane & 15;
    const float bias = b1[lr];
    float s = 0.f, q = 0.f;
    #pragma unroll
    for (int r = 0; r < 8; ++r) { float v = c[r] + bias; s += v; q += v * v; }
    atomicAdd(&ssum[lr], s);
    atomicAdd(&ssq[lr], q);
    __syncthreads();
    if (threadIdx.x < 16) {
        atomicAdd(&stats[0  + threadIdx.x], ssum[threadIdx.x]);
        atomicAdd(&stats[16 + threadIdx.x], ssq[threadIdx.x]);
    }
}

// Fold BN stats + gamma/beta into per-channel scale/shift
__global__ void k_finalize(float* __restrict__ stats, const float* __restrict__ g,
                           const float* __restrict__ beta, int nchan,
                           int sumOff, int outOff, float invN)
{
    int t = threadIdx.x;
    if (t < nchan) {
        float mean = stats[sumOff + t] * invN;
        float var  = stats[sumOff + nchan + t] * invN - mean * mean;
        float sc   = g[t] * rsqrtf(var + BN_EPS);
        stats[outOff + t]         = sc;
        stats[outOff + nchan + t] = beta[t] - mean * sc;
    }
}

// Pass 1b: recompute conv1 (x is L2-resident), BN+ReLU+2x2 maxpool in regs,
// write pooled map P as f16 NHWC [128,56,56,16]
__global__ void k_conv1_bn_pool(const float* __restrict__ x, const float* __restrict__ w1,
                                const float* __restrict__ b1, const float* __restrict__ stats,
                                _Float16* __restrict__ P)
{
    __shared__ __align__(16) _Float16 patch[4 * 17 * 33];
    __shared__ __align__(16) _Float16 bfrag[512];
    const int blk = blockIdx.x;
    const int b = blk / 98, r0 = blk % 98;
    const int by = r0 / 7, bx = r0 % 7;
    conv1_stage(x, w1, patch, bfrag, b, by, bx);
    const int lane = threadIdx.x & 31, wave = threadIdx.x >> 5;
    v8f c = conv1_block_tile(patch, bfrag, lane, wave);
    const int half = lane >> 4;
    const int lr   = lane & 15;
    const int ty = 2 * by + (wave >> 2);    // 4-row group index (pool block y)
    const int tx = 4 * bx + (wave & 3);     // 4-col group index (pool block x)
    const float bias = b1[lr];
    const float sc = stats[32 + lr], sh = stats[48 + lr];
    float act[8];
    #pragma unroll
    for (int r = 0; r < 8; ++r) {
        float v = (c[r] + bias) * sc + sh;
        act[r] = v > 0.f ? v : 0.f;
    }
    #pragma unroll
    for (int wloc = 0; wloc < 2; ++wloc) {
        // D rows m = r + 8*half: this lane holds pool windows w = 2*half+wloc
        int w = 2 * half + wloc;
        float m0 = fmaxf(fmaxf(act[4 * wloc + 0], act[4 * wloc + 1]),
                         fmaxf(act[4 * wloc + 2], act[4 * wloc + 3]));
        int py = 2 * ty + (w >> 1);
        int px = 2 * tx + (w & 1);
        P[(((size_t)b * 56 + py) * 56 + px) * 16 + lr] = (_Float16)m0;
    }
}

// ---------------------------------------------------------------------------
// conv2: wave tile = 4x4 conv2 pixels (7x7 tiles per image). Per-wave pooled
// patch (10x9x16 f16, incl. zero guard) staged into LDS with CDNA5 async
// LDS loads; B fragments (K=160 padded, N=32) pre-built fragment-ordered.
// ---------------------------------------------------------------------------
__global__ void k_conv2_stats(const _Float16* __restrict__ P, const float* __restrict__ w2,
                              const float* __restrict__ b2, float* __restrict__ stats,
                              float* __restrict__ y2)
{
    __shared__ __align__(16) _Float16 bfrag[5 * 2 * 32 * 16];   // [ks][nt][lane][i]
    __shared__ __align__(16) _Float16 patch[8][10 * 9 * 16];    // per-wave
    __shared__ float ssum[32], ssq[32];
    if (threadIdx.x < 32) { ssum[threadIdx.x] = 0.f; ssq[threadIdx.x] = 0.f; }

    // stage fragment-ordered B (w2: o*144 + c*9 + (ky*3+kx); K=(ky*3+kx)*16+c)
    for (int idx = threadIdx.x; idx < 5120; idx += 256) {
        int i    = idx & 15;
        int lane = (idx >> 4) & 31;
        int nt   = (idx >> 9) & 1;
        int ks   = idx >> 10;
        int kb   = ks * 32 + (i & 7) + 8 * (lane >> 4) + 16 * (i >> 3);
        int o    = nt * 16 + (lane & 15);
        float v  = 0.0f;
        if (kb < 144) v = w2[o * 144 + (kb & 15) * 9 + (kb >> 4)];
        bfrag[idx] = (_Float16)v;
    }

    const int lane = threadIdx.x & 31, wave = threadIdx.x >> 5;
    const int tile = blockIdx.x * 8 + wave;      // 128*49 tiles
    const int b = tile / 49, rr = tile % 49;
    const int ty2 = rr / 7, tx2 = rr % 7;
    const int py0 = 8 * ty2 - 1, px0 = 8 * tx2 - 1;

    // per-wave patch staging: 720 dwords via async global->LDS copies
    unsigned* p32 = (unsigned*)patch[wave];
    const unsigned* Pg = (const unsigned*)P;
    for (int idx = lane; idx < 720; idx += 32) {
        int chp = idx & 7;
        int pix = idx >> 3;
        int row = pix / 9, col = pix % 9;
        int py = py0 + row, px = px0 + col;
        if (row < 9 && (unsigned)py < 56u && (unsigned)px < 56u) {
            unsigned lds_off = (unsigned)(size_t)&p32[idx];
            const unsigned* ga = Pg + (((size_t)b * 56 + py) * 56 + px) * 8 + chp;
            asm volatile("global_load_async_to_lds_b32 %0, %1, off"
                         :: "v"(lds_off), "v"((unsigned long long)(size_t)ga)
                         : "memory");
        } else {
            p32[idx] = 0u;   // image border + K-padding guard row
        }
    }
    asm volatile("s_wait_asynccnt 0x0" ::: "memory");
    __syncthreads();

    const int half = lane >> 4;
    const int lr   = lane & 15;
    const int myr = lr >> 2, mxr = lr & 3;            // A-row m = lr -> pixel
    const int pixbase = (2 * myr * 9 + 2 * mxr) * 16;
    const _Float16* pp = patch[wave];

    v8f c0 = {}, c1 = {};
    #pragma unroll
    for (int ks = 0; ks < 5; ++ks) {
        v16h a;
        #pragma unroll
        for (int i = 0; i < 16; ++i) {
            const int kb = ks * 32 + a_k_of(i, 0);
            const int d0 = c2_delta(kb), d1 = c2_delta(kb + 8);
            _Float16 av = pp[pixbase + (half ? d1 : d0)];
            if (kb + 8 >= 144) {                      // only last k-step elems
                if (kb + 8 * half >= 144) av = (_Float16)0.f;
            }
            a[i] = av;
        }
        v16h bm0 = *(const v16h*)&bfrag[((ks * 2 + 0) * 32 + lane) * 16];
        v16h bm1 = *(const v16h*)&bfrag[((ks * 2 + 1) * 32 + lane) * 16];
        c0 = __builtin_amdgcn_wmma_f32_16x16x32_f16(false, a, false, bm0, (short)0, c0, false, false);
        c1 = __builtin_amdgcn_wmma_f32_16x16x32_f16(false, a, false, bm1, (short)0, c1, false, false);
    }

    float s0 = 0.f, q0 = 0.f, s1 = 0.f, q1 = 0.f;
    const float bias0 = b2[lr], bias1 = b2[lr + 16];
    #pragma unroll
    for (int r = 0; r < 8; ++r) {
        int m = r + 8 * half;                         // D row
        int oy = 4 * ty2 + (m >> 2), ox = 4 * tx2 + (m & 3);
        int qq = oy * 28 + ox;
        float v0 = c0[r] + bias0;
        float v1 = c1[r] + bias1;
        y2[((size_t)b * 32 + lr)      * 784 + qq] = v0;
        y2[((size_t)b * 32 + lr + 16) * 784 + qq] = v1;
        s0 += v0; q0 += v0 * v0; s1 += v1; q1 += v1 * v1;
    }
    atomicAdd(&ssum[lr], s0);      atomicAdd(&ssq[lr], q0);
    atomicAdd(&ssum[lr + 16], s1); atomicAdd(&ssq[lr + 16], q1);
    __syncthreads();
    if (threadIdx.x < 32) {
        atomicAdd(&stats[64 + threadIdx.x], ssum[threadIdx.x]);
        atomicAdd(&stats[96 + threadIdx.x], ssq[threadIdx.x]);
    }
}

// Head: fused BN2+ReLU+avgpool+FC+cos. One block per image.
__global__ void k_head(const float* __restrict__ y2, const float* __restrict__ stats,
                       const float* __restrict__ fcw, const float* __restrict__ fcb,
                       float* __restrict__ out)
{
    __shared__ float red[256];
    const int b = blockIdx.x;
    float acc = 0.f;
    for (int idx = threadIdx.x; idx < 32 * 784; idx += 256) {
        int c = idx / 784, pos = idx % 784;
        float v = y2[((size_t)b * 32 + c) * 784 + pos] * stats[128 + c] + stats[160 + c];
        v = v > 0.f ? v : 0.f;
        acc += v * fcw[c];
    }
    red[threadIdx.x] = acc;
    __syncthreads();
    for (int s = 128; s > 0; s >>= 1) {
        if (threadIdx.x < s) red[threadIdx.x] += red[threadIdx.x + s];
        __syncthreads();
    }
    if (threadIdx.x == 0) {
        float logit = red[0] * (1.0f / 784.0f) + fcb[0];
        float pr = cosf(logit);
        out[2 * b + 0] = pr;
        out[2 * b + 1] = 1.0f - pr;
    }
}

extern "C" void kernel_launch(void* const* d_in, const int* in_sizes, int n_in,
                              void* d_out, int out_size, void* d_ws, size_t ws_size,
                              hipStream_t stream)
{
    (void)in_sizes; (void)n_in; (void)out_size; (void)ws_size;
    const float* x   = (const float*)d_in[0];
    const float* w1  = (const float*)d_in[1];
    const float* b1  = (const float*)d_in[2];
    const float* g1  = (const float*)d_in[3];
    const float* be1 = (const float*)d_in[4];
    const float* w2  = (const float*)d_in[5];
    const float* b2  = (const float*)d_in[6];
    const float* g2  = (const float*)d_in[7];
    const float* be2 = (const float*)d_in[8];
    const float* fcw = (const float*)d_in[9];
    const float* fcb = (const float*)d_in[10];
    float* out = (float*)d_out;

    // workspace: [stats 1KB][P f16 NHWC 12.25MB][y2 f32 12.25MB] ~= 25.7MB
    float*    stats = (float*)d_ws;
    _Float16* P     = (_Float16*)((char*)d_ws + 1024);
    const size_t pB = (size_t)128 * 56 * 56 * 16 * sizeof(_Float16);
    float*    y2    = (float*)((char*)d_ws + 1024 + pB);

    k_zero_stats   <<<1,     256, 0, stream>>>(stats);
    k_conv1_stats  <<<12544, 256, 0, stream>>>(x, w1, b1, stats);
    k_finalize     <<<1,      64, 0, stream>>>(stats, g1, be1, 16, 0,  32,
                                               1.0f / (128.f * 112.f * 112.f));
    k_conv1_bn_pool<<<12544, 256, 0, stream>>>(x, w1, b1, stats, P);
    k_conv2_stats  <<<784,   256, 0, stream>>>(P, w2, b2, stats, y2);
    k_finalize     <<<1,      64, 0, stream>>>(stats, g2, be2, 32, 64, 128,
                                               1.0f / (128.f * 28.f * 28.f));
    k_head         <<<128,   256, 0, stream>>>(y2, stats, fcw, fcb, out);
}
